// GraphModule_61460982005897
// MI455X (gfx1250) — matrix operand order repaired and checked by reading, not compiled, not run on hardware
//
#include <hip/hip_runtime.h>

#define NN 1048576
#define NG 16384
#define DD 128

typedef _Float16 v16h __attribute__((ext_vector_type(16)));
typedef _Float16 v8h  __attribute__((ext_vector_type(8)));
typedef _Float16 v4h  __attribute__((ext_vector_type(4)));
typedef float    v8f  __attribute__((ext_vector_type(8)));
typedef int      v4i  __attribute__((ext_vector_type(4)));

#if __has_builtin(__builtin_amdgcn_global_load_async_to_lds_b128) && \
    __has_builtin(__builtin_amdgcn_global_store_async_from_lds_b128) && \
    __has_builtin(__builtin_amdgcn_s_wait_asynccnt)
#define HAVE_ASYNC 1
typedef __attribute__((address_space(1))) v4i as1_v4i;
typedef __attribute__((address_space(3))) v4i as3_v4i;
#else
#define HAVE_ASYNC 0
#endif

// ---- workspace layout (bytes) ----
static constexpr size_t H_OFF    = 0;                                    // h f16 [NN*DD]
static constexpr size_t S1_OFF   = (size_t)NN * DD * 2;                  // s1 f32 [NN]
static constexpr size_t S2_OFF   = S1_OFF + (size_t)NN * 4;              // s2 f32 [NN]
static constexpr size_t ZERO_OFF = S2_OFF + (size_t)NN * 4;              // zeroed region start
static constexpr size_t M1_OFF   = ZERO_OFF;                             // m1 enc u32 [NG]
static constexpr size_t M2_OFF   = M1_OFF + (size_t)NG * 4;
static constexpr size_t Z1_OFF   = M2_OFF + (size_t)NG * 4;
static constexpr size_t Z2_OFF   = Z1_OFF + (size_t)NG * 4;
static constexpr size_t CNT_OFF  = Z2_OFF + (size_t)NG * 4;
static constexpr size_t SUMH_OFF = CNT_OFF + (size_t)NG * 4;             // [NG*DD] f32
static constexpr size_t S1H_OFF  = SUMH_OFF + (size_t)NG * DD * 4;
static constexpr size_t S2H_OFF  = S1H_OFF + (size_t)NG * DD * 4;
static constexpr size_t ZERO_BYTES = (size_t)NG * 4 * 5 + (size_t)NG * DD * 4 * 3;
static constexpr int    ZERO_DW4   = (int)(ZERO_BYTES / 16);

// order-preserving float<->uint encoding for atomic max
__device__ __forceinline__ unsigned encf(float x) {
  unsigned b = __float_as_uint(x);
  return (b & 0x80000000u) ? ~b : (b | 0x80000000u);
}
__device__ __forceinline__ float decf(unsigned u) {
  unsigned b = (u & 0x80000000u) ? (u & 0x7FFFFFFFu) : ~u;
  return __uint_as_float(b);
}

__global__ void k_zero(uint4* __restrict__ p, int n4) {
  int i = blockIdx.x * blockDim.x + threadIdx.x;
  if (i < n4) { uint4 z; z.x = z.y = z.z = z.w = 0u; p[i] = z; }
}

// 64 rows per block, 4 waves; each wave computes a 16x128 stripe of h via WMMA.
__global__ __launch_bounds__(128)
void k_encoder(const float* __restrict__ x, const float* __restrict__ W,
               const float* __restrict__ be, const float* __restrict__ wq1,
               const float* __restrict__ bq1, const float* __restrict__ wq2,
               const float* __restrict__ bq2, const int* __restrict__ gid,
               _Float16* __restrict__ hws, float* __restrict__ s1ws,
               float* __restrict__ s2ws, unsigned* __restrict__ m1e,
               unsigned* __restrict__ m2e)
{
  __shared__ _Float16 ldsA[64 * 136];    // x tile [row][k] f16, pad 136; reused for h tile
  __shared__ _Float16 ldsW[128 * 136];   // W^T [n][k] f16, pad 136

  const int tid = threadIdx.x;
  const int lane = tid & 31;
  const int wv = tid >> 5;
  const int blockRow = blockIdx.x * 64;

  // ---- stage A tile: 64x128 f32 -> f16 ----
  const float4* xv = (const float4*)(x + (size_t)blockRow * DD);
#pragma unroll
  for (int it = 0; it < 16; ++it) {
    int idx = tid + it * 128;          // 2048 float4
    int r = idx >> 5, c4 = idx & 31;
    float4 v = xv[idx];
    v4h h4;
    h4[0] = (_Float16)v.x; h4[1] = (_Float16)v.y;
    h4[2] = (_Float16)v.z; h4[3] = (_Float16)v.w;
    *(v4h*)&ldsA[r * 136 + c4 * 4] = h4;
  }
  // ---- stage W transposed: Wt[n][k] = W[k*128+n], f32 -> f16 ----
  const float4* wv4 = (const float4*)W;
#pragma unroll
  for (int it = 0; it < 32; ++it) {
    int idx = tid + it * 128;          // 4096 float4
    int k = idx >> 5, n4 = (idx & 31) * 4;
    float4 v = wv4[idx];
    ldsW[(n4 + 0) * 136 + k] = (_Float16)v.x;
    ldsW[(n4 + 1) * 136 + k] = (_Float16)v.y;
    ldsW[(n4 + 2) * 136 + k] = (_Float16)v.z;
    ldsW[(n4 + 3) * 136 + k] = (_Float16)v.w;
  }
  __syncthreads();

  const int hrow = lane & 15;
  const int hi = lane >> 4;
  const int rowBase = wv * 16;

  v8f acc[8];
  const v8f vzero = {0.f, 0.f, 0.f, 0.f, 0.f, 0.f, 0.f, 0.f};
#pragma unroll
  for (int nt = 0; nt < 8; ++nt) acc[nt] = vzero;

#pragma unroll
  for (int kf = 0; kf < 4; ++kf) {
    // A fragment: lane -> row (lane&15); halves K = [k0,k0+8) and [k0+16,k0+24)
    const _Float16* arow = &ldsA[(rowBase + hrow) * 136 + kf * 32 + hi * 8];
    v8h alo = *(const v8h*)arow;
    v8h ahi = *(const v8h*)(arow + 16);
    v16h a = __builtin_shufflevector(alo, ahi, 0, 1, 2, 3, 4, 5, 6, 7,
                                     8, 9, 10, 11, 12, 13, 14, 15);
#pragma unroll
    for (int nt = 0; nt < 8; ++nt) {
      // B fragment: lane -> column n = nt*16 + (lane&15); halves K = [k0,k0+8) and [k0+16,k0+24)
      const _Float16* brow = &ldsW[(nt * 16 + hrow) * 136 + kf * 32 + hi * 8];
      v8h blo = *(const v8h*)brow;
      v8h bhi = *(const v8h*)(brow + 16);
      v16h b = __builtin_shufflevector(blo, bhi, 0, 1, 2, 3, 4, 5, 6, 7,
                                       8, 9, 10, 11, 12, 13, 14, 15);
      acc[nt] = __builtin_amdgcn_wmma_f32_16x16x32_f16(
          false, a, false, b, (short)0, acc[nt], false, false);
    }
  }

  __syncthreads();   // everyone done reading ldsA fragments; reuse as h tile

  // ---- bias + relu; re-tile h into LDS; fold score dot-products ----
  float p1[8] = {0, 0, 0, 0, 0, 0, 0, 0};
  float p2[8] = {0, 0, 0, 0, 0, 0, 0, 0};
#pragma unroll
  for (int nt = 0; nt < 8; ++nt) {
    int n = nt * 16 + hrow;
    float bn = be[n], w1n = wq1[n], w2n = wq2[n];
#pragma unroll
    for (int v = 0; v < 8; ++v) {
      float hv = fmaxf(acc[nt][v] + bn, 0.f);
      int ml = rowBase + v + hi * 8;            // local row 0..63
      ldsA[ml * 136 + n] = (_Float16)hv;
      p1[v] = fmaf(hv, w1n, p1[v]);
      p2[v] = fmaf(hv, w2n, p2[v]);
    }
  }
  __syncthreads();

  // ---- coalesced 128-bit write-back of the h tile ----
#pragma unroll
  for (int it = 0; it < 8; ++it) {
    int idx = tid + it * 128;                   // 1024 x 16B chunks
    int r = idx >> 4, c = idx & 15;
    char* gaddr = (char*)hws + (size_t)(blockRow + r) * (DD * 2) + c * 16;
    char* laddr = (char*)&ldsA[0] + r * 272 + c * 16;
#if HAVE_ASYNC
    __builtin_amdgcn_global_store_async_from_lds_b128(
        (as1_v4i*)gaddr, (as3_v4i*)laddr, 0, 0);
#else
    *(v8h*)gaddr = *(const v8h*)laddr;
#endif
  }

  // ---- per-row scores: reduce over each 16-lane half ----
  float bq1v = bq1[0], bq2v = bq2[0];
#pragma unroll
  for (int v = 0; v < 8; ++v) {
#pragma unroll
    for (int msk = 1; msk < 16; msk <<= 1) {
      p1[v] += __shfl_xor(p1[v], msk);
      p2[v] += __shfl_xor(p2[v], msk);
    }
    if (hrow == v) {
      int r = blockRow + rowBase + v + hi * 8;
      float s1v = p1[v] + bq1v;
      float s2v = p2[v] + bq2v;
      s1ws[r] = s1v;
      s2ws[r] = s2v;
      int g = gid[r];
      atomicMax(&m1e[g], encf(s1v));
      atomicMax(&m2e[g], encf(s2v));
    }
  }
  // async stores drain at S_ENDPGM (implicit wait-idle)
}

// sorted-segment accumulation: one wave per 256-node chunk, 4 features per lane,
// register accumulation per run of equal graph_id, atomic flush at boundaries.
__global__ __launch_bounds__(64)
void k_accum(const _Float16* __restrict__ h, const float* __restrict__ s1,
             const float* __restrict__ s2, const int* __restrict__ gid,
             const unsigned* __restrict__ m1e, const unsigned* __restrict__ m2e,
             float* __restrict__ z1, float* __restrict__ z2,
             float* __restrict__ cnt, float* __restrict__ sumH,
             float* __restrict__ s1H, float* __restrict__ s2H)
{
#if HAVE_ASYNC
  __shared__ _Float16 hbuf[2 * 2 * 32 * DD];   // 2 waves x double-buffer x 32 nodes
#endif
  const int lane = threadIdx.x & 31;
  const int wv = threadIdx.x >> 5;
  const size_t start = ((size_t)blockIdx.x * 2 + wv) * 256;
  const int fbase = lane * 4;

  int curG = gid[start];
  float aH[4] = {0, 0, 0, 0}, a1[4] = {0, 0, 0, 0}, a2[4] = {0, 0, 0, 0};
  float az1 = 0.f, az2 = 0.f, ac = 0.f;

  auto flush = [&](int g) {
#pragma unroll
    for (int c = 0; c < 4; ++c) {
      atomicAdd(&sumH[(size_t)g * DD + fbase + c], aH[c]);
      atomicAdd(&s1H[(size_t)g * DD + fbase + c], a1[c]);
      atomicAdd(&s2H[(size_t)g * DD + fbase + c], a2[c]);
      aH[c] = a1[c] = a2[c] = 0.f;
    }
    if (lane == 0) {
      atomicAdd(&z1[g], az1);
      atomicAdd(&z2[g], az2);
      atomicAdd(&cnt[g], ac);
      az1 = az2 = ac = 0.f;
    }
  };
  auto body = [&](int g, float s1v, float s2v, v4h h4) {
    if (g != curG) { flush(curG); curG = g; }
    float e1 = __expf(s1v - decf(m1e[g]));
    float e2 = __expf(s2v - decf(m2e[g]));
#pragma unroll
    for (int c = 0; c < 4; ++c) {
      float hv = (float)h4[c];
      aH[c] += hv;
      a1[c] = fmaf(e1, hv, a1[c]);
      a2[c] = fmaf(e2, hv, a2[c]);
    }
    if (lane == 0) { az1 += e1; az2 += e2; ac += 1.f; }
  };

#if HAVE_ASYNC
  _Float16* mybuf = &hbuf[wv * 2 * 32 * DD];
  auto issue = [&](int batch, int bsel) {
    const char* gb = (const char*)(h + (start + (size_t)batch * 32) * DD);
    char* lb = (char*)(mybuf + bsel * 32 * DD);
#pragma unroll
    for (int j = 0; j < 16; ++j) {   // 32 nodes * 256B = 16 x (32 lanes x 16B)
      __builtin_amdgcn_global_load_async_to_lds_b128(
          (as1_v4i*)(gb + j * 512 + lane * 16),
          (as3_v4i*)(lb + j * 512 + lane * 16), 0, 0);
    }
  };
  issue(0, 0);
  for (int b = 0; b < 8; ++b) {
    if (b + 1 < 8) {
      issue(b + 1, (b + 1) & 1);
      __builtin_amdgcn_s_wait_asynccnt(16);   // batch b complete (in-order)
    } else {
      __builtin_amdgcn_s_wait_asynccnt(0);
    }
    const _Float16* lbase = mybuf + (b & 1) * 32 * DD;
    for (int i = 0; i < 32; i += 4) {
      size_t node = start + (size_t)b * 32 + i;
      int4   g4 = *(const int4*)&gid[node];
      float4 sa = *(const float4*)&s1[node];
      float4 sb = *(const float4*)&s2[node];
      v4h h0 = *(const v4h*)&lbase[(i + 0) * DD + fbase];
      v4h h1 = *(const v4h*)&lbase[(i + 1) * DD + fbase];
      v4h h2 = *(const v4h*)&lbase[(i + 2) * DD + fbase];
      v4h h3 = *(const v4h*)&lbase[(i + 3) * DD + fbase];
      body(g4.x, sa.x, sb.x, h0);
      body(g4.y, sa.y, sb.y, h1);
      body(g4.z, sa.z, sb.z, h2);
      body(g4.w, sa.w, sb.w, h3);
    }
  }
#else
  for (int i = 0; i < 256; i += 4) {
    size_t node = start + i;
    int4   g4 = *(const int4*)&gid[node];
    float4 sa = *(const float4*)&s1[node];
    float4 sb = *(const float4*)&s2[node];
    v4h h0 = *(const v4h*)&h[(node + 0) * DD + fbase];
    v4h h1 = *(const v4h*)&h[(node + 1) * DD + fbase];
    v4h h2 = *(const v4h*)&h[(node + 2) * DD + fbase];
    v4h h3 = *(const v4h*)&h[(node + 3) * DD + fbase];
    body(g4.x, sa.x, sb.x, h0);
    body(g4.y, sa.y, sb.y, h1);
    body(g4.z, sa.z, sb.z, h2);
    body(g4.w, sa.w, sb.w, h3);
  }
#endif
  flush(curG);
}

__global__ void k_final(const float* __restrict__ sumH, const float* __restrict__ s1H,
                        const float* __restrict__ s2H, const float* __restrict__ z1,
                        const float* __restrict__ z2, const float* __restrict__ cnt,
                        float* __restrict__ out)
{
  int idx = blockIdx.x * blockDim.x + threadIdx.x;   // [0, NG*DD)
  if (idx >= NG * DD) return;
  int g = idx >> 7;
  const int GD = NG * DD;
  out[idx]          = sumH[idx] / fmaxf(cnt[g], 1.f);
  out[GD + idx]     = s1H[idx] / fmaxf(z1[g], 1e-12f);
  out[2 * GD + idx] = s2H[idx] / fmaxf(z2[g], 1e-12f);
}

extern "C" void kernel_launch(void* const* d_in, const int* in_sizes, int n_in,
                              void* d_out, int out_size, void* d_ws, size_t ws_size,
                              hipStream_t stream) {
  const float* x   = (const float*)d_in[0];
  const int*   gid = (const int*)d_in[1];
  const float* W   = (const float*)d_in[2];
  const float* be  = (const float*)d_in[3];
  const float* wq1 = (const float*)d_in[4];
  const float* bq1 = (const float*)d_in[5];
  const float* wq2 = (const float*)d_in[6];
  const float* bq2 = (const float*)d_in[7];
  char* ws = (char*)d_ws;

  _Float16* hws = (_Float16*)(ws + H_OFF);
  float*    s1  = (float*)(ws + S1_OFF);
  float*    s2  = (float*)(ws + S2_OFF);
  unsigned* m1e = (unsigned*)(ws + M1_OFF);
  unsigned* m2e = (unsigned*)(ws + M2_OFF);
  float*    z1  = (float*)(ws + Z1_OFF);
  float*    z2  = (float*)(ws + Z2_OFF);
  float*    cnt = (float*)(ws + CNT_OFF);
  float*    sumH = (float*)(ws + SUMH_OFF);
  float*    s1H  = (float*)(ws + S1H_OFF);
  float*    s2H  = (float*)(ws + S2H_OFF);
  float*    out  = (float*)d_out;

  k_zero<<<(ZERO_DW4 + 255) / 256, 256, 0, stream>>>((uint4*)(ws + ZERO_OFF), ZERO_DW4);
  k_encoder<<<NN / 64, 128, 0, stream>>>(x, W, be, wq1, bq1, wq2, bq2, gid,
                                         hws, s1, s2, m1e, m2e);
  k_accum<<<NN / 512, 64, 0, stream>>>(hws, s1, s2, gid, m1e, m2e,
                                       z1, z2, cnt, sumH, s1H, s2H);
  k_final<<<(NG * DD) / 256, 256, 0, stream>>>(sumH, s1H, s2H, z1, z2, cnt, out);
}
